// GATModel_76647986364937
// MI455X (gfx1250) — matrix operand order repaired and checked by reading, not compiled, not run on hardware
//
#include <hip/hip_runtime.h>
#include <math.h>

#define NNODES 50000
#define NEDGES 800000
#define NGRAPH 512
#define INDIM  32
#define HIDC   100
#define HP     112     // HID padded to 7*16 (WMMA N tiles)
#define K2P    128     // layer-2 K (100) padded to 4*32 (WMMA K chunks)
#define NEG_SLOPE 0.2f

typedef __attribute__((ext_vector_type(16))) _Float16 v16h;
typedef __attribute__((ext_vector_type(8)))  float    v8f;

// ---------------- WMMA fragment helpers (wave32 layouts, CDNA5 ISA) ----
// 16-bit A 16x32: lane L holds row M=L%16; elems 0..7 = K kb+8*hi+0..7,
//                 elems 8..15 = K kb+8*hi+16..23  (hi = L>=16)
__device__ __forceinline__ v16h load_a_frag(const _Float16* rowp, int kbase, int laneHi) {
    union { v16h v; float4 f4[2]; } u;
    u.f4[0] = *(const float4*)(rowp + kbase + 8 * laneHi);
    u.f4[1] = *(const float4*)(rowp + kbase + 8 * laneHi + 16);
    return u.v;
}

// Prepacked B fragment: contiguous 16 halves per lane -> one 32B vector load.
__device__ __forceinline__ v16h load_b_packed(const _Float16* __restrict__ Wp,
                                              int nt, int kc, int nk, int lane) {
    return *(const v16h*)(Wp + (((size_t)nt * nk + kc) * 32 + (size_t)lane) * 16);
}

// ---------------- weight prepack: W[kreal x 100] f32 -> B-fragment-ordered f16 ----
// elem index i = ((nt*nk + kc)*32 + lane)*16 + e ; B layout: K = kc*32+8*hi+e (e<8), kc*32+16+8*hi+e-8 (e>=8), N = nt*16+lane%16
__global__ void k_pack_b(const float* __restrict__ W, int kreal, int nk,
                         _Float16* __restrict__ Wp) {
    const int total = 7 * nk * 32 * 16;
    for (int i = blockIdx.x * blockDim.x + threadIdx.x; i < total;
         i += gridDim.x * blockDim.x) {
        int e    = i & 15;
        int lane = (i >> 4) & 31;
        int g    = i >> 9;            // nt*nk + kc
        int kc   = g % nk;
        int nt   = g / nk;
        int k    = kc * 32 + 8 * (lane >> 4) + (e < 8 ? e : e + 8);
        int col  = nt * 16 + (lane & 15);
        float wv = (col < HIDC && k < kreal) ? W[k * HIDC + col] : 0.0f;
        Wp[i] = (_Float16)wv;
    }
}

// pad a length-100 vector to 112 with zeros
__global__ void k_pad112(const float* __restrict__ src, float* __restrict__ dst) {
    int i = threadIdx.x;
    if (i < HP) dst[i] = (i < HIDC) ? src[i] : 0.0f;
}

// ---------------- GEMM: Out[M x HP] = A_f16[M x lda] @ Wp + biasP --------------
// one wave per 16-row tile, full 112-col stripe (7 accumulators); block = 8 waves
__global__ void k_gemm_bias(const _Float16* __restrict__ A, int lda,
                            const _Float16* __restrict__ Wp,
                            const float* __restrict__ biasP,   // [112], zero padded
                            float* __restrict__ Out, int M) {
    const int lane   = threadIdx.x & 31;
    const int wave   = threadIdx.x >> 5;
    const int laneHi = lane >> 4;
    const int m16    = lane & 15;
    const int tileM  = blockIdx.x * 8 + wave;
    if (tileM * 16 >= M) return;               // wave-uniform exit: EXEC full at WMMA
    int row = tileM * 16 + m16;
    if (row >= M) row = M - 1;
    const _Float16* rowp = A + (size_t)row * lda;
    const int nk = lda >> 5;
    v8f zero = {};
    v8f acc[7];
#pragma unroll
    for (int nt = 0; nt < 7; ++nt) acc[nt] = zero;
    for (int kc = 0; kc < nk; ++kc) {
        v16h a = load_a_frag(rowp, kc * 32, laneHi);
#pragma unroll
        for (int nt = 0; nt < 7; ++nt) {
            v16h b = load_b_packed(Wp, nt, kc, nk, lane);
            acc[nt] = __builtin_amdgcn_wmma_f32_16x16x32_f16(false, a, false, b,
                                                             (short)0, acc[nt],
                                                             false, false);
        }
    }
#pragma unroll
    for (int nt = 0; nt < 7; ++nt) {
        int col = nt * 16 + m16;
        float bv = biasP[col];
#pragma unroll
        for (int v = 0; v < 8; ++v) {          // C layout: VGPR v -> row v+8*hi, N=lane%16
            int m = tileM * 16 + v + 8 * laneHi;
            if (m < M) Out[(size_t)m * HP + col] = acc[nt][v] + bv;
        }
    }
}

// ---------------- Fused edge kernel: logits[e] = att . lrelu(xl[src]+xr[dst]+ea@We)
// one wave per 16-edge tile; block = 8 waves; e_emb stays in WMMA accumulators.
__global__ void k_edge_logits(const _Float16* __restrict__ ea,   // [E x 32] f16
                              const _Float16* __restrict__ WeP,  // packed, nk=1
                              const float* __restrict__ attP,    // [112], zero padded
                              const float* __restrict__ xl,      // [N x HP]
                              const float* __restrict__ xr,      // [N x HP]
                              const int* __restrict__ srcI,
                              const int* __restrict__ dstI,
                              float* __restrict__ logits) {
    const int lane   = threadIdx.x & 31;
    const int wave   = threadIdx.x >> 5;
    const int laneHi = lane >> 4;
    const int m16    = lane & 15;
    const int tile   = blockIdx.x * 8 + wave;  // NEDGES/16 tiles, exact multiple of 8
    const _Float16* rowp = ea + (size_t)(tile * 16 + m16) * INDIM;
    v16h a = load_a_frag(rowp, 0, laneHi);

    int sv[8], dv[8];
#pragma unroll
    for (int v = 0; v < 8; ++v) {              // edge ids this lane's acc rows map to
        int e = tile * 16 + v + 8 * laneHi;
        sv[v] = srcI[e];
        dv[v] = dstI[e];
    }
    float partial[8] = {0.f,0.f,0.f,0.f,0.f,0.f,0.f,0.f};
#pragma unroll
    for (int nt = 0; nt < 7; ++nt) {
        v16h b = load_b_packed(WeP, nt, 0, 1, lane);
        v8f acc = {};
        acc = __builtin_amdgcn_wmma_f32_16x16x32_f16(false, a, false, b,
                                                     (short)0, acc, false, false);
        int col = nt * 16 + m16;
        float av = attP[col];                  // zero kills padded cols
#pragma unroll
        for (int v = 0; v < 8; ++v) {
            float z = acc[v] + xl[(size_t)sv[v] * HP + col] + xr[(size_t)dv[v] * HP + col];
            z = (z > 0.f) ? z : NEG_SLOPE * z;
            partial[v] += av * z;
        }
    }
#pragma unroll
    for (int v = 0; v < 8; ++v) {              // reduce the 16 lanes of each half-wave
        float s = partial[v];
        s += __shfl_xor(s, 1);
        s += __shfl_xor(s, 2);
        s += __shfl_xor(s, 4);
        s += __shfl_xor(s, 8);
        partial[v] = s;
    }
    if (m16 == 0) {
#pragma unroll
        for (int v = 0; v < 8; ++v)
            logits[tile * 16 + v + 8 * laneHi] = partial[v];
    }
}

// ---------------- elementwise / segment kernels -----------------------------------
__global__ void k_fill(float* __restrict__ p, float v, long n) {
    long i = (long)blockIdx.x * blockDim.x + threadIdx.x;
    long st = (long)gridDim.x * blockDim.x;
    for (; i < n; i += st) p[i] = v;
}

__global__ void k_cvt_f16(const float* __restrict__ in, _Float16* __restrict__ out, long n) {
    long i = (long)blockIdx.x * blockDim.x + threadIdx.x;
    long st = (long)gridDim.x * blockDim.x;
    for (; i < n; i += st) out[i] = (_Float16)in[i];
}

__device__ __forceinline__ void atomicMaxFloat(float* addr, float val) {
    if (val >= 0.0f) atomicMax((int*)addr, __float_as_int(val));
    else             atomicMin((unsigned int*)addr, __float_as_uint(val));
}

__global__ void k_segmax(const float* __restrict__ logits, const int* __restrict__ dstI,
                         float* __restrict__ segmax, int E) {
    int i = blockIdx.x * blockDim.x + threadIdx.x;
    if (i < E) atomicMaxFloat(&segmax[dstI[i]], logits[i]);
}

__global__ void k_expsum(float* __restrict__ logits, const int* __restrict__ dstI,
                         const float* __restrict__ segmax, float* __restrict__ segsum, int E) {
    int i = blockIdx.x * blockDim.x + threadIdx.x;
    if (i < E) {
        int d = dstI[i];
        float a = __expf(logits[i] - segmax[d]);
        logits[i] = a;                          // overwrite with unnormalized alpha
        atomicAdd(&segsum[d], a);
    }
}

__global__ void k_init_bias(float* __restrict__ h, const float* __restrict__ bias, int M) {
    long n = (long)M * HP;
    long i = (long)blockIdx.x * blockDim.x + threadIdx.x;
    long st = (long)gridDim.x * blockDim.x;
    for (; i < n; i += st) {
        int c = (int)(i % HP);
        h[i] = (c < HIDC) ? bias[c] : 0.0f;
    }
}

// one wave per edge: h[dst] += (alpha/denom) * xl[src]
__global__ void k_aggregate(const float* __restrict__ alpha,
                            const int* __restrict__ srcI, const int* __restrict__ dstI,
                            const float* __restrict__ segsum,
                            const float* __restrict__ xl,
                            float* __restrict__ h, int E) {
    int wid  = (int)(((long)blockIdx.x * blockDim.x + threadIdx.x) >> 5);
    int lane = threadIdx.x & 31;
    if (wid >= E) return;
    int s = srcI[wid], d = dstI[wid];
    float w = alpha[wid] / (segsum[d] + 1e-16f);
    for (int c = lane; c < HIDC; c += 32)
        atomicAdd(&h[(size_t)d * HP + c], w * xl[(size_t)s * HP + c]);
}

__global__ void k_relu_cvt(const float* __restrict__ h, _Float16* __restrict__ out, int M) {
    long n = (long)M * K2P;
    long i = (long)blockIdx.x * blockDim.x + threadIdx.x;
    long st = (long)gridDim.x * blockDim.x;
    for (; i < n; i += st) {
        int c = (int)(i % K2P);
        float v = 0.0f;
        if (c < HIDC) {
            v = h[(i / K2P) * HP + c];
            v = v > 0.0f ? v : 0.0f;
        }
        out[i] = (_Float16)v;
    }
}

// one wave per node: pooled[batch[n]] += h2[n]; cnt[batch[n]] += 1
__global__ void k_pool(const float* __restrict__ h2, const int* __restrict__ batch,
                       float* __restrict__ pooled, float* __restrict__ cnt, int M) {
    int wid  = (int)(((long)blockIdx.x * blockDim.x + threadIdx.x) >> 5);
    int lane = threadIdx.x & 31;
    if (wid >= M) return;
    int b = batch[wid];
    for (int c = lane; c < HIDC; c += 32)
        atomicAdd(&pooled[(size_t)b * HIDC + c], h2[(size_t)wid * HP + c]);
    if (lane == 0) atomicAdd(&cnt[b], 1.0f);
}

__global__ void k_final(const float* __restrict__ pooled, const float* __restrict__ cnt,
                        const float* __restrict__ Wlin, const float* __restrict__ blin,
                        float* __restrict__ out, int G) {
    int g = blockIdx.x * blockDim.x + threadIdx.x;
    if (g >= G) return;
    float cv = cnt[g];
    cv = cv > 1.0f ? cv : 1.0f;
    float acc = 0.0f;
    for (int c = 0; c < HIDC; ++c)
        acc += (pooled[(size_t)g * HIDC + c] / cv) * Wlin[c];
    acc += blin[0];
    out[g] = 1.0f / (1.0f + __expf(-acc));
}

// ---------------- launch ----------------------------------------------------------
extern "C" void kernel_launch(void* const* d_in, const int* in_sizes, int n_in,
                              void* d_out, int out_size, void* d_ws, size_t ws_size,
                              hipStream_t stream) {
    (void)in_sizes; (void)n_in; (void)out_size; (void)ws_size;
    const float* x     = (const float*)d_in[0];
    const int*   eidx  = (const int*)d_in[1];    // [2 x E] int32
    const float* eattr = (const float*)d_in[2];
    const int*   batch = (const int*)d_in[3];
    const float* Wl1 = (const float*)d_in[4];
    const float* bl1 = (const float*)d_in[5];
    const float* Wr1 = (const float*)d_in[6];
    const float* br1 = (const float*)d_in[7];
    const float* We1 = (const float*)d_in[8];
    const float* at1 = (const float*)d_in[9];
    const float* b1  = (const float*)d_in[10];
    const float* Wl2 = (const float*)d_in[11];
    const float* bl2 = (const float*)d_in[12];
    const float* Wr2 = (const float*)d_in[13];
    const float* br2 = (const float*)d_in[14];
    const float* We2 = (const float*)d_in[15];
    const float* at2 = (const float*)d_in[16];
    const float* b2  = (const float*)d_in[17];
    const float* Wln = (const float*)d_in[18];
    const float* bln = (const float*)d_in[19];
    const int* srcI = eidx;
    const int* dstI = eidx + NEDGES;

    // workspace carve (all 256B aligned)
    char* w = (char*)d_ws;
    size_t off = 0;
    auto carve = [&](size_t bytes) -> void* {
        void* p = w + off;
        off = (off + bytes + 255) & ~(size_t)255;
        return p;
    };
    float*    xl     = (float*)   carve((size_t)NNODES * HP * 4);
    float*    xr     = (float*)   carve((size_t)NNODES * HP * 4);
    float*    hacc   = (float*)   carve((size_t)NNODES * HP * 4);   // layer1 out, reused for layer2
    _Float16* xf16   = (_Float16*)carve((size_t)NNODES * INDIM * 2);
    _Float16* hf16   = (_Float16*)carve((size_t)NNODES * K2P * 2);
    _Float16* eaf16  = (_Float16*)carve((size_t)NEDGES * INDIM * 2);
    float*    logits = (float*)   carve((size_t)NEDGES * 4);
    float*    segmax = (float*)   carve((size_t)NNODES * 4);
    float*    segsum = (float*)   carve((size_t)NNODES * 4);
    float*    pooled = (float*)   carve((size_t)NGRAPH * HIDC * 4);
    float*    cnt    = (float*)   carve((size_t)NGRAPH * 4);
    // prepacked B-matrices (f16, fragment order) and padded vectors
    _Float16* pWl1 = (_Float16*)carve((size_t)7 * 1 * 512 * 2);
    _Float16* pWr1 = (_Float16*)carve((size_t)7 * 1 * 512 * 2);
    _Float16* pWe1 = (_Float16*)carve((size_t)7 * 1 * 512 * 2);
    _Float16* pWl2 = (_Float16*)carve((size_t)7 * 4 * 512 * 2);
    _Float16* pWr2 = (_Float16*)carve((size_t)7 * 4 * 512 * 2);
    _Float16* pWe2 = (_Float16*)carve((size_t)7 * 1 * 512 * 2);
    float* pbl1 = (float*)carve(HP * 4);
    float* pbr1 = (float*)carve(HP * 4);
    float* pat1 = (float*)carve(HP * 4);
    float* pbl2 = (float*)carve(HP * 4);
    float* pbr2 = (float*)carve(HP * 4);
    float* pat2 = (float*)carve(HP * 4);

    const int gemmBlocks = (NNODES / 16 + 7) / 8;     // 8 M-tiles per 256-thr block
    const int edgeBlocks = (NEDGES / 16) / 8;         // 50000 tiles / 8
    const int eThreads   = (NEDGES + 255) / 256;

    // ---- prep: pack weights into WMMA-B order, pad vectors, convert activations ----
    k_pack_b<<<14, 256, 0, stream>>>(Wl1, INDIM, 1, pWl1);
    k_pack_b<<<14, 256, 0, stream>>>(Wr1, INDIM, 1, pWr1);
    k_pack_b<<<14, 256, 0, stream>>>(We1, INDIM, 1, pWe1);
    k_pack_b<<<56, 256, 0, stream>>>(Wl2, HIDC, 4, pWl2);
    k_pack_b<<<56, 256, 0, stream>>>(Wr2, HIDC, 4, pWr2);
    k_pack_b<<<14, 256, 0, stream>>>(We2, INDIM, 1, pWe2);
    k_pad112<<<1, 128, 0, stream>>>(bl1, pbl1);
    k_pad112<<<1, 128, 0, stream>>>(br1, pbr1);
    k_pad112<<<1, 128, 0, stream>>>(at1, pat1);
    k_pad112<<<1, 128, 0, stream>>>(bl2, pbl2);
    k_pad112<<<1, 128, 0, stream>>>(br2, pbr2);
    k_pad112<<<1, 128, 0, stream>>>(at2, pat2);
    k_cvt_f16<<<6250, 256, 0, stream>>>(x, xf16, (long)NNODES * INDIM);
    k_cvt_f16<<<100000, 256, 0, stream>>>(eattr, eaf16, (long)NEDGES * INDIM);

    // ---- layer 1 ----
    k_gemm_bias<<<gemmBlocks, 256, 0, stream>>>(xf16, INDIM, pWl1, pbl1, xl, NNODES);
    k_gemm_bias<<<gemmBlocks, 256, 0, stream>>>(xf16, INDIM, pWr1, pbr1, xr, NNODES);
    k_fill<<<196, 256, 0, stream>>>(segmax, -INFINITY, (long)NNODES);
    k_fill<<<196, 256, 0, stream>>>(segsum, 0.0f, (long)NNODES);
    k_edge_logits<<<edgeBlocks, 256, 0, stream>>>(eaf16, pWe1, pat1, xl, xr, srcI, dstI, logits);
    k_segmax<<<eThreads, 256, 0, stream>>>(logits, dstI, segmax, NEDGES);
    k_expsum<<<eThreads, 256, 0, stream>>>(logits, dstI, segmax, segsum, NEDGES);
    k_init_bias<<<21875, 256, 0, stream>>>(hacc, b1, NNODES);
    k_aggregate<<<NEDGES / 8, 256, 0, stream>>>(logits, srcI, dstI, segsum, xl, hacc, NEDGES);

    // relu + f16 (K padded to 128 with zeros)
    k_relu_cvt<<<25000, 256, 0, stream>>>(hacc, hf16, NNODES);

    // ---- layer 2 ----
    k_gemm_bias<<<gemmBlocks, 256, 0, stream>>>(hf16, K2P, pWl2, pbl2, xl, NNODES);
    k_gemm_bias<<<gemmBlocks, 256, 0, stream>>>(hf16, K2P, pWr2, pbr2, xr, NNODES);
    k_fill<<<196, 256, 0, stream>>>(segmax, -INFINITY, (long)NNODES);
    k_fill<<<196, 256, 0, stream>>>(segsum, 0.0f, (long)NNODES);
    k_edge_logits<<<edgeBlocks, 256, 0, stream>>>(eaf16, pWe2, pat2, xl, xr, srcI, dstI, logits);
    k_segmax<<<eThreads, 256, 0, stream>>>(logits, dstI, segmax, NEDGES);
    k_expsum<<<eThreads, 256, 0, stream>>>(logits, dstI, segmax, segsum, NEDGES);
    k_init_bias<<<21875, 256, 0, stream>>>(hacc, b2, NNODES);
    k_aggregate<<<NEDGES / 8, 256, 0, stream>>>(logits, srcI, dstI, segsum, xl, hacc, NEDGES);

    // ---- pool + head ----
    k_fill<<<200, 256, 0, stream>>>(pooled, 0.0f, (long)NGRAPH * HIDC);
    k_fill<<<2, 256, 0, stream>>>(cnt, 0.0f, (long)NGRAPH);
    k_pool<<<(NNODES * 32 + 255) / 256, 256, 0, stream>>>(hacc, batch, pooled, cnt, NNODES);
    k_final<<<(NGRAPH + 255) / 256, 256, 0, stream>>>(pooled, cnt, Wln, bln, (float*)d_out, NGRAPH);
}
